// HybridKANFFN_51934744543466
// MI455X (gfx1250) — matrix operand order repaired and checked by reading, not compiled, not run on hardware
//
#include <hip/hip_runtime.h>

// ---------------- problem constants (from reference) ----------------
#define DD   512      // model dim
#define DHH  128      // expert hidden dim
#define TT   64       // tiles (experts)
#define CC   8        // clusters
#define TPCC 8        // tiles per cluster
#define GG   16       // spline grid
#define NN   2048     // tokens (B*S = 2*1024)
#define EPSF 1e-5f

typedef __attribute__((ext_vector_type(16))) __bf16     v16bf;
typedef __attribute__((ext_vector_type(8)))  float      v8f;
typedef __attribute__((ext_vector_type(4)))  unsigned   v4u;
typedef __attribute__((ext_vector_type(4)))  int        v4i;

union FragBF { v16bf bf; v4u q[2]; };

// async global->LDS copy path (CDNA5 GLOBAL_LOAD_ASYNC_TO_LDS_B128, ASYNCcnt)
#if __has_builtin(__builtin_amdgcn_global_load_async_to_lds_b128) && \
    __has_builtin(__builtin_amdgcn_s_wait_asynccnt)
#define USE_ASYNC_LDS 1
#define AS1 __attribute__((address_space(1)))
#define AS3 __attribute__((address_space(3)))
#else
#define USE_ASYNC_LDS 0
#endif

__device__ __forceinline__ unsigned short f2bf(float f) {
    unsigned u = __float_as_uint(f);
    u += 0x7FFFu + ((u >> 16) & 1u);      // round-to-nearest-even
    return (unsigned short)(u >> 16);
}

// ---------------- 1: zero counters ----------------
__global__ void k_init(int* counts, int* cursor) {
    int i = threadIdx.x;
    if (i < TT) { counts[i] = 0; cursor[i] = 0; }
}

// ---------------- 2: weights -> bf16 ----------------
__global__ void k_convert(const float* __restrict__ Wd, const float* __restrict__ Wu,
                          unsigned short* __restrict__ wd_bf, unsigned short* __restrict__ wu_bf) {
    size_t i = (size_t)blockIdx.x * blockDim.x + threadIdx.x;
    if (i >= (size_t)TT * DHH * DD) return;
    wd_bf[i] = f2bf(Wd[i]);
    wu_bf[i] = f2bf(Wu[i]);
}

// ---------------- 3: tile signatures: sign(mean_h Wd[t,h,d]) ----------------
__global__ void k_tile_sigs(const float* __restrict__ Wd, float* __restrict__ tsig) {
    int t = blockIdx.x;
    for (int d = threadIdx.x; d < DD; d += blockDim.x) {
        float s = 0.f;
        for (int h = 0; h < DHH; h++)
            s += Wd[((size_t)t * DHH + h) * DD + d];
        tsig[(size_t)t * DD + d] = (s > 0.f) ? 1.f : ((s < 0.f) ? -1.f : 0.f);
    }
}

// ---------------- 4: cluster signatures ----------------
__global__ void k_cluster_sigs(const float* __restrict__ tsig, float* __restrict__ csig) {
    int i = blockIdx.x * blockDim.x + threadIdx.x;   // c*DD + d
    if (i >= CC * DD) return;
    int c = i / DD, d = i % DD;
    float s = 0.f;
    for (int j = 0; j < TPCC; j++)
        s += tsig[(size_t)(c * TPCC + j) * DD + d];
    csig[i] = (s > 0.f) ? 1.f : ((s < 0.f) ? -1.f : 0.f);
}

// ---------------- 5: LayerNorm + hierarchical routing (1 block / token) ----------------
__global__ void k_ln_route(const float* __restrict__ x, const float* __restrict__ gamma,
                           const float* __restrict__ beta, const float* __restrict__ tsig,
                           const float* __restrict__ csig, unsigned short* __restrict__ xn_bf,
                           int* __restrict__ tidx, int* __restrict__ counts) {
    int n = blockIdx.x, tid = threadIdx.x;
    const float* xr = x + (size_t)n * DD;
    __shared__ float red[256];
    __shared__ float sc[CC];
    __shared__ float ts[TPCC];
    __shared__ int   s_best;

    float v0 = xr[tid], v1 = xr[tid + 256];

    red[tid] = v0 + v1;
    __syncthreads();
    for (int s = 128; s > 0; s >>= 1) { if (tid < s) red[tid] += red[tid + s]; __syncthreads(); }
    float mu = red[0] * (1.0f / DD);
    __syncthreads();

    float d0 = v0 - mu, d1 = v1 - mu;
    red[tid] = d0 * d0 + d1 * d1;
    __syncthreads();
    for (int s = 128; s > 0; s >>= 1) { if (tid < s) red[tid] += red[tid + s]; __syncthreads(); }
    float rinv = rsqrtf(red[0] * (1.0f / DD) + EPSF);
    __syncthreads();

    float xn0 = d0 * rinv * gamma[tid] + beta[tid];
    float xn1 = d1 * rinv * gamma[tid + 256] + beta[tid + 256];
    xn_bf[(size_t)n * DD + tid]       = f2bf(xn0);
    xn_bf[(size_t)n * DD + tid + 256] = f2bf(xn1);
    float s0 = (xn0 > 0.f) ? 1.f : ((xn0 < 0.f) ? -1.f : 0.f);
    float s1 = (xn1 > 0.f) ? 1.f : ((xn1 < 0.f) ? -1.f : 0.f);

    // cluster scores (sums of +/-1 -> float atomics are exact & deterministic)
    if (tid < CC) sc[tid] = 0.f;
    __syncthreads();
#pragma unroll
    for (int c = 0; c < CC; c++) {
        float p = s0 * csig[c * DD + tid] + s1 * csig[c * DD + tid + 256];
        atomicAdd(&sc[c], p);
    }
    __syncthreads();
    if (tid == 0) {
        int b = 0; float bv = sc[0];
        for (int c = 1; c < CC; c++) if (sc[c] > bv) { bv = sc[c]; b = c; }
        s_best = b;
    }
    __syncthreads();
    int tb = s_best * TPCC;
    if (tid < TPCC) ts[tid] = 0.f;
    __syncthreads();
#pragma unroll
    for (int j = 0; j < TPCC; j++) {
        float p = s0 * tsig[(size_t)(tb + j) * DD + tid] + s1 * tsig[(size_t)(tb + j) * DD + tid + 256];
        atomicAdd(&ts[j], p);
    }
    __syncthreads();
    if (tid == 0) {
        int b = 0; float bv = ts[0];
        for (int j = 1; j < TPCC; j++) if (ts[j] > bv) { bv = ts[j]; b = j; }
        int t = tb + b;
        tidx[n] = t;
        atomicAdd(&counts[t], 1);
    }
}

// ---------------- 6: exclusive scan over 64 counts ----------------
__global__ void k_scan(const int* __restrict__ counts, int* __restrict__ offsets) {
    if (threadIdx.x == 0) {
        int run = 0;
        for (int t = 0; t < TT; t++) { offsets[t] = run; run += counts[t]; }
    }
}

// ---------------- 7: scatter tokens into per-expert lists ----------------
__global__ void k_scatter(const int* __restrict__ tidx, const int* __restrict__ offsets,
                          int* __restrict__ cursor, int* __restrict__ order) {
    int n = blockIdx.x * blockDim.x + threadIdx.x;
    if (n >= NN) return;
    int t = tidx[n];
    int p = atomicAdd(&cursor[t], 1);
    order[offsets[t] + p] = n;
}

// ---------------- 8: fused expert kernel: GEMM1 -> spline -> GEMM2 -> residual ----------------
// One block per expert tile; 8 waves (256 threads, wave32).
__global__ void __launch_bounds__(256)
k_moe(const float* __restrict__ x, const unsigned short* __restrict__ xn_bf,
      const unsigned short* __restrict__ wd_bf, const unsigned short* __restrict__ wu_bf,
      const float* __restrict__ sbase, const float* __restrict__ sslope,
      const float* __restrict__ scale, const int* __restrict__ order,
      const int* __restrict__ offsets, const int* __restrict__ counts,
      float* __restrict__ out) {
    int t = blockIdx.x;
    int cnt = counts[t];
    if (cnt == 0) return;
    int off = offsets[t];

    __shared__ float          s_base[DHH * GG];    // 8 KB spline bases for this expert
    __shared__ float          s_slope[DHH * GG];   // 8 KB
    __shared__ unsigned short s_hs[16 * DHH];      // 4 KB bf16 hidden activations
    __shared__ int            s_tok[16];

    // ---- stage spline tables into LDS (async global->LDS when available) ----
#if USE_ASYNC_LDS
    {
        const char* gb = (const char*)(sbase  + (size_t)t * DHH * GG);
        const char* gs = (const char*)(sslope + (size_t)t * DHH * GG);
        char* lb = (char*)s_base;
        char* ls = (char*)s_slope;
        for (int i = threadIdx.x * 16; i < DHH * GG * 4; i += blockDim.x * 16) {
            __builtin_amdgcn_global_load_async_to_lds_b128(
                (AS1 v4i*)(unsigned long long)(gb + i),
                (AS3 v4i*)(unsigned)(unsigned long long)(lb + i), 0, 0);
            __builtin_amdgcn_global_load_async_to_lds_b128(
                (AS1 v4i*)(unsigned long long)(gs + i),
                (AS3 v4i*)(unsigned)(unsigned long long)(ls + i), 0, 0);
        }
        __builtin_amdgcn_s_wait_asynccnt(0);
    }
#else
    for (int i = threadIdx.x; i < DHH * GG; i += blockDim.x) {
        s_base[i]  = sbase[(size_t)t * DHH * GG + i];
        s_slope[i] = sslope[(size_t)t * DHH * GG + i];
    }
#endif
    float sc_t = scale[t];

    int lane = threadIdx.x & 31;
    int wave = threadIdx.x >> 5;     // 0..7
    int half = lane >> 4;            // 0/1
    int l15  = lane & 15;

    for (int chunk = 0; chunk < cnt; chunk += 16) {
        int nvalid = cnt - chunk; if (nvalid > 16) nvalid = 16;
        __syncthreads();                       // protect s_tok / s_hs reuse
        if (threadIdx.x < 16) {
            int idx = chunk + threadIdx.x;
            s_tok[threadIdx.x] = order[off + (idx < cnt ? idx : chunk)];  // pad tail
        }
        __syncthreads();

        // ---- GEMM1: h[16 tok x 16 dh], dh-tile = wave; K = 512 in 16 wmma steps ----
        int dh0 = wave * 16;
        int mtokA = s_tok[l15];                // A layout: both lane halves carry row M=l15
        const v4u* arow = (const v4u*)(xn_bf + (size_t)mtokA * DD);
        const v4u* brow = (const v4u*)(wd_bf + ((size_t)t * DHH + dh0 + l15) * DD);
        v8f acc = {0.f,0.f,0.f,0.f,0.f,0.f,0.f,0.f};
#pragma unroll 4
        for (int kb = 0; kb < DD; kb += 32) {
            FragBF a, b;
            a.q[0] = arow[(kb + half * 8) >> 3];
            a.q[1] = arow[(kb + 16 + half * 8) >> 3];
            int bk = (kb + half * 8) >> 3;
            b.q[0] = brow[bk];
            b.q[1] = brow[bk + 2];
            __builtin_prefetch(brow + bk + 8, 0, 3);   // stream next weight chunk (near caches)
            acc = __builtin_amdgcn_wmma_f32_16x16x32_bf16(
                      false, a.bf, false, b.bf, (short)0, acc, false, false);
        }

        // ---- spline nonlinearity on C fragment: lane holds (m = v+8*half, n = l15) ----
#pragma unroll
        for (int v = 0; v < 8; v++) {
            int m  = v + half * 8;
            int dh = dh0 + l15;
            float h  = acc[v];
            // fast sigmoid: v_exp + v_rcp (1-ulp) -- gives spline cell and in-cell offset
            float hn = __builtin_amdgcn_rcpf(1.f + __expf(-h));
            float g  = hn * (float)GG;
            int gi = (int)g;
            gi = gi < 0 ? 0 : (gi > GG - 1 ? GG - 1 : gi);
            float hs = s_base[dh * GG + gi] + s_slope[dh * GG + gi] * (g - (float)gi);
            s_hs[m * DHH + dh] = f2bf(hs);
        }
        __syncthreads();

        // ---- GEMM2: y[16 tok x 512], each wave does 4 d-tiles; K = 128 in 4 steps ----
        for (int dt = 0; dt < 4; dt++) {
            int d0 = (wave * 4 + dt) * 16;
            const v4u* ar = (const v4u*)(s_hs + l15 * DHH);
            const v4u* br = (const v4u*)(wu_bf + ((size_t)t * DD + d0 + l15) * DHH);
            v8f acc2 = {0.f,0.f,0.f,0.f,0.f,0.f,0.f,0.f};
#pragma unroll
            for (int kb = 0; kb < DHH; kb += 32) {
                FragBF a, b;
                a.q[0] = ar[(kb + half * 8) >> 3];          // ds_load_b128 from LDS
                a.q[1] = ar[(kb + 16 + half * 8) >> 3];
                int bk = (kb + half * 8) >> 3;
                b.q[0] = br[bk];
                b.q[1] = br[bk + 2];
                acc2 = __builtin_amdgcn_wmma_f32_16x16x32_bf16(
                           false, a.bf, false, b.bf, (short)0, acc2, false, false);
            }
#pragma unroll
            for (int v = 0; v < 8; v++) {
                int m = v + half * 8;
                if (m < nvalid) {
                    int tok = s_tok[m];
                    size_t o = (size_t)tok * DD + d0 + l15;
                    out[o] = x[o] + acc2[v] * sc_t;        // residual + per-tile scale
                }
            }
        }
    }
}

// ---------------- host launch ----------------
extern "C" void kernel_launch(void* const* d_in, const int* in_sizes, int n_in,
                              void* d_out, int out_size, void* d_ws, size_t ws_size,
                              hipStream_t stream) {
    (void)in_sizes; (void)n_in; (void)out_size; (void)ws_size;
    const float* x     = (const float*)d_in[0];
    const float* gamma = (const float*)d_in[1];
    const float* beta  = (const float*)d_in[2];
    const float* Wd    = (const float*)d_in[3];
    const float* sbase = (const float*)d_in[4];
    const float* sslop = (const float*)d_in[5];
    const float* Wu    = (const float*)d_in[6];
    const float* scale = (const float*)d_in[7];
    float* out = (float*)d_out;

    char* ws = (char*)d_ws;
    size_t off = 0;
    auto take = [&](size_t nbytes) -> void* {
        void* p = ws + off;
        off += (nbytes + 255) & ~(size_t)255;
        return p;
    };
    unsigned short* wd_bf  = (unsigned short*)take((size_t)TT * DHH * DD * 2);
    unsigned short* wu_bf  = (unsigned short*)take((size_t)TT * DD * DHH * 2);
    unsigned short* xn_bf  = (unsigned short*)take((size_t)NN * DD * 2);
    float*          tsig   = (float*)take((size_t)TT * DD * 4);
    float*          csig   = (float*)take((size_t)CC * DD * 4);
    int*            tidx   = (int*)take((size_t)NN * 4);
    int*            counts = (int*)take((size_t)TT * 4);
    int*            offs   = (int*)take((size_t)TT * 4);
    int*            cursor = (int*)take((size_t)TT * 4);
    int*            order  = (int*)take((size_t)NN * 4);

    k_init<<<1, 64, 0, stream>>>(counts, cursor);

    size_t nconv = (size_t)TT * DHH * DD;
    k_convert<<<(unsigned)((nconv + 255) / 256), 256, 0, stream>>>(Wd, Wu, wd_bf, wu_bf);

    k_tile_sigs<<<TT, 256, 0, stream>>>(Wd, tsig);
    k_cluster_sigs<<<(CC * DD + 255) / 256, 256, 0, stream>>>(tsig, csig);

    k_ln_route<<<NN, 256, 0, stream>>>(x, gamma, beta, tsig, csig, xn_bf, tidx, counts);

    k_scan<<<1, 64, 0, stream>>>(counts, offs);
    k_scatter<<<(NN + 255) / 256, 256, 0, stream>>>(tidx, offs, cursor, order);

    k_moe<<<TT, 256, 0, stream>>>(x, xn_bf, wd_bf, wu_bf, sbase, sslop, scale,
                                  order, offs, counts, out);
}
